// DeepEquiCategorySpecificMLP_35442070126939
// MI455X (gfx1250) — matrix operand order repaired and checked by reading, not compiled, not run on hardware
//
#include <hip/hip_runtime.h>
#include <hip/hip_bf16.h>

// ---------------------------------------------------------------------------
// Problem constants (from reference): N=2048 rows, D=256, H=1024, C=8 experts
// ---------------------------------------------------------------------------
#define N_SAMPLES 2048
#define DIM       256
#define HID       1024
#define NCAT      8
#define TILE_M    32                  // rows per GEMM tile (2 WMMA row-tiles)
#define NP        2304                // padded rows: 2048 + 8*32 (32-aligned segs)
#define MAXT      (NP / TILE_M)       // 72 tiles max
#define EPS_LN    1e-5f

typedef __attribute__((ext_vector_type(16))) __bf16 v16bf;
typedef __attribute__((ext_vector_type(8)))  float  v8f;

// float -> bf16 with round-to-nearest-even (used outside GEMM hot loop)
static __device__ inline __bf16 f2bf(float f) {
    union { float f; unsigned u; } v; v.f = f;
    unsigned r = v.u + 0x7FFFu + ((v.u >> 16) & 1u);
    union { unsigned short s; __bf16 b; } o; o.s = (unsigned short)(r >> 16);
    return o.b;
}

// pack two fp32 -> bf16 pair (truncate) with a single v_perm_b32
static __device__ inline unsigned pack_bf2_trunc(float lo, float hi) {
    union { float f; unsigned u; } a, b; a.f = lo; b.f = hi;
    // result b0<-a.b2, b1<-a.b3, b2<-b.b2, b3<-b.b3  (src1=a, src0=b)
    return __builtin_amdgcn_perm(b.u, a.u, 0x07060302u);
}

// ---------------------------------------------------------------------------
// 1) per-category histogram
// ---------------------------------------------------------------------------
__global__ void khist(const int* __restrict__ cat, int* __restrict__ counts) {
    int i = blockIdx.x * blockDim.x + threadIdx.x;
    if (i < N_SAMPLES) atomicAdd(&counts[cat[i]], 1);
}

// ---------------------------------------------------------------------------
// 2) build 32-aligned category segments + 32-row tile table (C=8, trivial)
// ---------------------------------------------------------------------------
__global__ void kscan(const int* __restrict__ counts, int* __restrict__ cursors,
                      int* __restrict__ tileCat, int* __restrict__ tileRow,
                      int* __restrict__ totalTiles) {
    if (threadIdx.x != 0 || blockIdx.x != 0) return;
    int off = 0, t = 0;
    for (int c = 0; c < NCAT; ++c) {
        cursors[c] = off;
        int nt = (counts[c] + TILE_M - 1) / TILE_M;
        for (int i = 0; i < nt; ++i) {
            tileCat[t] = c; tileRow[t] = off + i * TILE_M; ++t;
        }
        off += nt * TILE_M;
    }
    totalTiles[0] = t;
}

// ---------------------------------------------------------------------------
// 3) input LayerNorm + bf16 convert + scatter into category segment
// ---------------------------------------------------------------------------
__global__ __launch_bounds__(256) void kgather(const float* __restrict__ x,
                                               const int* __restrict__ cat,
                                               int* __restrict__ cursors,
                                               int* __restrict__ pos_of,
                                               __bf16* __restrict__ xg) {
    __shared__ float sm[512];
    __shared__ int spos;
    int row = blockIdx.x, t = threadIdx.x;
    float v = x[(size_t)row * DIM + t];
    sm[t] = v; sm[256 + t] = v * v; __syncthreads();
    for (int s = 128; s > 0; s >>= 1) {
        if (t < s) { sm[t] += sm[t + s]; sm[256 + t] += sm[256 + t + s]; }
        __syncthreads();
    }
    float mean = sm[0] * (1.0f / DIM);
    float var  = sm[256] * (1.0f / DIM) - mean * mean;
    float rstd = rsqrtf(var + EPS_LN);
    if (t == 0) {
        spos = atomicAdd(&cursors[cat[row]], 1);
        pos_of[row] = spos;
    }
    __syncthreads();
    int pos = spos;
    xg[(size_t)pos * DIM + t] = f2bf((v - mean) * rstd);
}

// ---------------------------------------------------------------------------
// 4) WMMA GEMM: out[m,n] = act( A[m,:] @ W[cat][:,n] + bias[cat][n] )
//    A: bf16 row-major [NP x K] (category-sorted); W: fp32 [C x K x NOUT]
//    block = 256 thr = 8 waves; wave -> 32x16 tile (2 WMMA sharing one B frag)
//    WG covers 32 rows x 128 cols.  K/NOUT compile-time -> immediate offsets.
// ---------------------------------------------------------------------------
template <int MODE, int K, int NOUT>
__global__ __launch_bounds__(256) void kgemm(const __bf16* __restrict__ A,
                                             const float* __restrict__ W,
                                             const float* __restrict__ bias,
                                             float* __restrict__ outf,
                                             __bf16* __restrict__ outb,
                                             const int* __restrict__ tileCat,
                                             const int* __restrict__ tileRow,
                                             const int* __restrict__ totalTiles) {
    int t = blockIdx.x;
    if (t >= totalTiles[0]) return;
    int cat = tileCat[t];
    int rowBase = tileRow[t];
    int lane = threadIdx.x & 31;
    int wave = threadIdx.x >> 5;
    int colBase = blockIdx.y * 128 + wave * 16;
    int n = colBase + (lane & 15);

    const float* Wn = W + (size_t)cat * K * NOUT + n;        // column base
    const unsigned short* Arow0 =
        (const unsigned short*)A + (size_t)(rowBase + (lane & 15)) * K;
    const unsigned short* Arow1 = Arow0 + (size_t)16 * K;    // rows +16
    int hiA = (lane & 16) ? 16 : 0;   // A: lane-halves K 0-15 vs 16-31
    int hiB = (lane & 16) ? 16 : 0;   // B: lane-halves K 0-15 vs 16-31

    v8f acc0 = {};
    v8f acc1 = {};
#pragma unroll 2
    for (int k0 = 0; k0 < K; k0 += 32) {
        union { v16bf v; uint4 q[2]; } a0, a1;
        union { v16bf v; unsigned u[8]; } bfr;
        int kb = k0 + hiA;
        a0.q[0] = *(const uint4*)(Arow0 + kb);        // K kb..kb+7
        a0.q[1] = *(const uint4*)(Arow0 + kb + 8);    // K kb+8..kb+15
        a1.q[0] = *(const uint4*)(Arow1 + kb);
        a1.q[1] = *(const uint4*)(Arow1 + kb + 8);
        const float* Wk = Wn + (size_t)(k0 + hiB) * NOUT;
#pragma unroll
        for (int v = 0; v < 8; ++v) {
            float f0 = Wk[(2 * v) * NOUT];
            float f1 = Wk[(2 * v + 1) * NOUT];
            bfr.u[v] = pack_bf2_trunc(f0, f1);
        }
        acc0 = __builtin_amdgcn_wmma_f32_16x16x32_bf16(false, a0.v, false, bfr.v,
                                                       (short)0, acc0, false, false);
        acc1 = __builtin_amdgcn_wmma_f32_16x16x32_bf16(false, a1.v, false, bfr.v,
                                                       (short)0, acc1, false, false);
    }

    float bv = bias[(size_t)cat * NOUT + n];
    int rowHi = (lane & 16) ? 8 : 0;  // C/D layout: VGPR r -> M = r (+8 hi lanes)
#pragma unroll
    for (int r = 0; r < 8; ++r) {
        int mr0 = rowBase + rowHi + r;
        int mr1 = mr0 + 16;
        float v0 = acc0[r] + bv;
        float v1 = acc1[r] + bv;
        if (MODE == 1) {
            v0 = v0 > 0.0f ? v0 : 0.0f;
            v1 = v1 > 0.0f ? v1 : 0.0f;
            outb[(size_t)mr0 * NOUT + n] = f2bf(v0);
            outb[(size_t)mr1 * NOUT + n] = f2bf(v1);
        } else {
            outf[(size_t)mr0 * NOUT + n] = v0;
            outf[(size_t)mr1 * NOUT + n] = v1;
        }
    }
}

// ---------------------------------------------------------------------------
// 5) g = LN(main * sigmoid(gate)) over H, write bf16.  One block per row.
// ---------------------------------------------------------------------------
__global__ __launch_bounds__(256) void kgateln(const float* __restrict__ M,
                                               const float* __restrict__ G,
                                               __bf16* __restrict__ out) {
    __shared__ float sm[512];
    int row = blockIdx.x, t = threadIdx.x;
    const float* mr = M + (size_t)row * HID;
    const float* gr = G + (size_t)row * HID;
    float g[4], s = 0.f, s2 = 0.f;
#pragma unroll
    for (int i = 0; i < 4; ++i) {
        int idx = t + 256 * i;
        float sg = 1.0f / (1.0f + __expf(-gr[idx]));
        float v = mr[idx] * sg;
        g[i] = v; s += v; s2 += v * v;
    }
    sm[t] = s; sm[256 + t] = s2; __syncthreads();
    for (int st = 128; st > 0; st >>= 1) {
        if (t < st) { sm[t] += sm[t + st]; sm[256 + t] += sm[256 + t + st]; }
        __syncthreads();
    }
    float mean = sm[0] * (1.0f / HID);
    float var  = sm[256] * (1.0f / HID) - mean * mean;
    float rstd = rsqrtf(var + EPS_LN);
    __bf16* orow = out + (size_t)row * HID;
#pragma unroll
    for (int i = 0; i < 4; ++i) orow[t + 256 * i] = f2bf((g[i] - mean) * rstd);
}

// ---------------------------------------------------------------------------
// 6) hidden LN over H, fp32 -> bf16.  One block per row.
// ---------------------------------------------------------------------------
__global__ __launch_bounds__(256) void kln(const float* __restrict__ X,
                                           __bf16* __restrict__ out) {
    __shared__ float sm[512];
    int row = blockIdx.x, t = threadIdx.x;
    const float* xr = X + (size_t)row * HID;
    float v[4], s = 0.f, s2 = 0.f;
#pragma unroll
    for (int i = 0; i < 4; ++i) {
        float x = xr[t + 256 * i];
        v[i] = x; s += x; s2 += x * x;
    }
    sm[t] = s; sm[256 + t] = s2; __syncthreads();
    for (int st = 128; st > 0; st >>= 1) {
        if (t < st) { sm[t] += sm[t + st]; sm[256 + t] += sm[256 + t + st]; }
        __syncthreads();
    }
    float mean = sm[0] * (1.0f / HID);
    float var  = sm[256] * (1.0f / HID) - mean * mean;
    float rstd = rsqrtf(var + EPS_LN);
    __bf16* orow = out + (size_t)row * HID;
#pragma unroll
    for (int i = 0; i < 4; ++i) orow[t + 256 * i] = f2bf((v[i] - mean) * rstd);
}

// ---------------------------------------------------------------------------
// 7) residual + output LN, scattered back to original row order.
// ---------------------------------------------------------------------------
__global__ __launch_bounds__(256) void kout(const float* __restrict__ Y,
                                            const float* __restrict__ x,
                                            const int* __restrict__ pos_of,
                                            float* __restrict__ out) {
    __shared__ float sm[512];
    int row = blockIdx.x, t = threadIdx.x;
    int pos = pos_of[row];
    float v = Y[(size_t)pos * DIM + t] + 0.1f * x[(size_t)row * DIM + t];
    sm[t] = v; sm[256 + t] = v * v; __syncthreads();
    for (int s = 128; s > 0; s >>= 1) {
        if (t < s) { sm[t] += sm[t + s]; sm[256 + t] += sm[256 + t + s]; }
        __syncthreads();
    }
    float mean = sm[0] * (1.0f / DIM);
    float var  = sm[256] * (1.0f / DIM) - mean * mean;
    float rstd = rsqrtf(var + EPS_LN);
    out[(size_t)row * DIM + t] = (v - mean) * rstd;
}

// ---------------------------------------------------------------------------
// launcher
// ---------------------------------------------------------------------------
extern "C" void kernel_launch(void* const* d_in, const int* in_sizes, int n_in,
                              void* d_out, int out_size, void* d_ws, size_t ws_size,
                              hipStream_t stream) {
    (void)in_sizes; (void)n_in; (void)out_size; (void)ws_size;
    const float* x   = (const float*)d_in[0];
    const int*   cat = (const int*)d_in[1];
    const float* W0  = (const float*)d_in[2];
    const float* b0  = (const float*)d_in[3];
    const float* Wm  = (const float*)d_in[4];
    const float* bm  = (const float*)d_in[5];
    const float* Wg  = (const float*)d_in[6];
    const float* bg  = (const float*)d_in[7];
    const float* Wog = (const float*)d_in[8];
    const float* bog = (const float*)d_in[9];
    const float* W2  = (const float*)d_in[10];
    const float* b2  = (const float*)d_in[11];
    float* out = (float*)d_out;

    char* w = (char*)d_ws;
    size_t off = 0;
    auto alloc = [&](size_t bytes) {
        size_t p = off;
        off += (bytes + 255) & ~(size_t)255;
        return p;
    };
    int*    counts  = (int*)(w + alloc(NCAT * 4));
    int*    cursors = (int*)(w + alloc(NCAT * 4));
    int*    tileCat = (int*)(w + alloc(MAXT * 4));
    int*    tileRow = (int*)(w + alloc(MAXT * 4));
    int*    totalT  = (int*)(w + alloc(4));
    int*    pos_of  = (int*)(w + alloc(N_SAMPLES * 4));
    __bf16* xg      = (__bf16*)(w + alloc((size_t)NP * DIM * 2));
    __bf16* H1b     = (__bf16*)(w + alloc((size_t)NP * HID * 2));
    float*  Amain   = (float*)(w + alloc((size_t)NP * HID * 4));
    float*  Agate   = (float*)(w + alloc((size_t)NP * HID * 4));
    __bf16* Gb      = (__bf16*)(w + alloc((size_t)NP * HID * 2));
    __bf16* H2b     = (__bf16*)(w + alloc((size_t)NP * HID * 2));
    float*  Yb      = (float*)(w + alloc((size_t)NP * DIM * 4));

    // reset per-call state (graph-capture safe)
    hipMemsetAsync(counts, 0, NCAT * 4, stream);
    hipMemsetAsync(xg, 0, (size_t)NP * DIM * 2, stream);   // zero pad rows

    khist<<<(N_SAMPLES + 255) / 256, 256, 0, stream>>>(cat, counts);
    kscan<<<1, 1, 0, stream>>>(counts, cursors, tileCat, tileRow, totalT);
    kgather<<<N_SAMPLES, 256, 0, stream>>>(x, cat, cursors, pos_of, xg);

    // h1 = relu(xn @ W0 + b0)                 [NP x 1024] bf16
    kgemm<1, DIM, HID><<<dim3(MAXT, HID / 128), 256, 0, stream>>>(
        xg, W0, b0, nullptr, H1b, tileCat, tileRow, totalT);
    // main = h1 @ Wm + bm ; gate = h1 @ Wg + bg   (fp32)
    kgemm<0, HID, HID><<<dim3(MAXT, HID / 128), 256, 0, stream>>>(
        H1b, Wm, bm, Amain, nullptr, tileCat, tileRow, totalT);
    kgemm<0, HID, HID><<<dim3(MAXT, HID / 128), 256, 0, stream>>>(
        H1b, Wg, bg, Agate, nullptr, tileCat, tileRow, totalT);
    // g = LN(main * sigmoid(gate))            [NP x 1024] bf16
    kgateln<<<NP, 256, 0, stream>>>(Amain, Agate, Gb);
    // h2 = g @ Wog + bog (fp32, reuse Amain)
    kgemm<0, HID, HID><<<dim3(MAXT, HID / 128), 256, 0, stream>>>(
        Gb, Wog, bog, Amain, nullptr, tileCat, tileRow, totalT);
    // h2n = LN(h2)                            [NP x 1024] bf16
    kln<<<NP, 256, 0, stream>>>(Amain, H2b);
    // y = h2n @ W2 + b2                       [NP x 256] fp32
    kgemm<0, HID, DIM><<<dim3(MAXT, DIM / 128), 256, 0, stream>>>(
        H2b, W2, b2, Yb, nullptr, tileCat, tileRow, totalT);
    // out = LN(y_scattered + 0.1 x)
    kout<<<N_SAMPLES, 256, 0, stream>>>(Yb, x, pos_of, out);
}
